// QREncoder_32263794327960
// MI455X (gfx1250) — matrix operand order repaired and checked by reading, not compiled, not run on hardware
//
#include <hip/hip_runtime.h>

// ---------------- types for WMMA ----------------
typedef __bf16 bf16;
typedef bf16  v16bf __attribute__((ext_vector_type(16)));
typedef float v8f   __attribute__((ext_vector_type(8)));

#define SIGC      4680          // 8 + 64 + 512 + 4096
#define KPAD      4704          // 147 * 32 (K padded for uniform WMMA loop)
#define SB_STRIDE 4712          // bf16 elems; (4712*2/4) % 64 == 52 == 4*13 -> conflict-free rows
#define ST_STRIDE 4688          // f32 state row stride
#define NSTEP     255
#define ODIM      512
#define LNUM      256
#define CIN       7
#define MROWS     32            // steps buffered per chunk (2 WMMA M-tiles)
#define NCHUNK    8             // 7*32 + 31 = 255

struct SharedMem {
  __attribute__((aligned(16))) bf16  sbuf[MROWS][SB_STRIDE]; // 301,568 B : bf16 A-tiles (32 steps x K)
  __attribute__((aligned(16))) float state[ST_STRIDE];       //  18,752 B : f32 signature state
  __attribute__((aligned(16))) float dxc[MROWS][8];          //   1,024 B : this chunk's increments
};                                                           // total 321,344 B < 320 KB WGP LDS

// ---- one-time W f32 -> bf16 K-padded copy into workspace: Wbf[512][KPAD] ----
__global__ __launch_bounds__(256)
void convert_w_kernel(const float* __restrict__ W, bf16* __restrict__ Wbf) {
  const int kk = blockIdx.x * 256 + threadIdx.x;   // gridDim.x = 19 -> covers 4704
  const int n  = blockIdx.y;                       // 512
  if (kk < KPAD)
    Wbf[(size_t)n * KPAD + kk] = (kk < SIGC) ? (bf16)W[(size_t)n * SIGC + kk] : (bf16)0.0f;
}

template <bool BF16W>
__global__ __launch_bounds__(256)
void sig_stream_fused_kernel(const float* __restrict__ inp,   // [64, 256, 7]
                             const float* __restrict__ W,     // [512, 4680] f32 (fallback path)
                             const bf16*  __restrict__ Wbf,   // [512, KPAD] bf16 (fast path)
                             const float* __restrict__ bias,  // [512]
                             float* __restrict__ out)         // [64, 255, 512]
{
  __shared__ SharedMem sm;
  const int tid = threadIdx.x;
  const int b   = blockIdx.x;
  const float* ib = inp + (size_t)b * LNUM * CIN;

  // ---------------- init: zero sbuf (incl. K-pad cols), zero state ----------------
  {
    unsigned* z = (unsigned*)&sm.sbuf[0][0];
    const int nz = MROWS * SB_STRIDE / 2;            // bf16 pairs as u32
    for (int i = tid; i < nz; i += 256) z[i] = 0u;
    for (int i = tid; i < ST_STRIDE; i += 256) sm.state[i] = 0.0f;
  }
  __syncthreads();

  const int lane = tid & 31;
  const int wv   = tid >> 5;        // 8 waves
  const int m16  = lane & 15;       // A row (within M-tile) / B column / C column lane
  const int hi   = lane >> 4;       // half-wave select
  const int n0w  = wv * 64;         // each wave owns 4 N-tiles of 16

  union V16 { uint4 u[2]; v16bf v; };
  const float dt = 1.0f / 255.0f;

  for (int chunk = 0; chunk < NCHUNK; ++chunk) {
    const int steps = (chunk == NCHUNK - 1) ? 31 : MROWS;

    // ---- stage this chunk's increments (time-augmented path differences) ----
    for (int q = tid; q < steps * 8; q += 256) {
      int m = q >> 3, c = q & 7, s = chunk * MROWS + m;
      sm.dxc[m][c] = (c == 0) ? dt : (ib[(s + 1) * CIN + (c - 1)] - ib[s * CIN + (c - 1)]);
    }
    __syncthreads();

    // ---------------- sequential signature scan: Chen's identity ----------------
    for (int m = 0; m < steps; ++m) {
      const float* __restrict__ P  = sm.state;
      const float* __restrict__ vv = sm.dxc[m];
      bf16* __restrict__ srow = sm.sbuf[m];
      float nv[19];

#pragma unroll
      for (int e = 0; e < 19; ++e) {
        const int idx = tid + 256 * e;
        if (idx < SIGC) {
          float r;
          if (idx < 8) {                       // level 1: a1 + v
            r = P[idx] + vv[idx];
          } else if (idx < 72) {               // level 2
            int j = idx - 8; int i1 = j >> 3, i2 = j & 7;
            float v1 = vv[i1], v2 = vv[i2];
            r = P[idx] + v1 * v2 * 0.5f + P[i1] * v2;
          } else if (idx < 584) {              // level 3
            int j = idx - 72;
            int i1 = j >> 6, i3 = j & 7;
            float v1 = vv[i1], v2 = vv[(j >> 3) & 7], v3 = vv[i3];
            float e2 = v2 * v3 * 0.5f;                  // e2(i2,i3)
            r = P[idx] + (v1 * e2) * (1.0f / 3.0f)      // e3 = v1*v2*v3/6
              + P[i1] * e2                              // a1 (x) e2
              + P[8 + (j >> 3)] * v3;                   // a2 (x) e1
          } else {                             // level 4
            int j = idx - 584;
            int i1 = j >> 9, i4 = j & 7;
            float v1 = vv[i1], v2 = vv[(j >> 6) & 7], v3 = vv[(j >> 3) & 7], v4 = vv[i4];
            float e2 = v3 * v4 * 0.5f;                  // e2(i3,i4)
            float e3 = v2 * e2 * (1.0f / 3.0f);         // e3(i2,i3,i4)
            float e4 = v1 * e3 * 0.25f;                 // e4 = v1v2v3v4/24
            r = P[idx] + e4
              + P[i1] * e3                              // a1 (x) e3
              + P[8  + (j >> 6)] * e2                   // a2 (x) e2
              + P[72 + (j >> 3)] * v4;                  // a3 (x) e1
          }
          nv[e] = r;
        }
      }
      __syncthreads();                                  // all reads of state done
#pragma unroll
      for (int e = 0; e < 19; ++e) {
        const int idx = tid + 256 * e;
        if (idx < SIGC) {
          sm.state[idx] = nv[e];                        // f32 recurrence state
          srow[idx]     = (bf16)nv[e];                  // bf16 GEMM operand
        }
      }
      __syncthreads();                                  // writes visible before next step
    }

    // ---------------- WMMA GEMM: [32 x KPAD]bf16 x [KPAD x 512]bf16 -> f32 ----------------
    const bf16* __restrict__ arow0 = sm.sbuf[m16];       // M-tile 0: rows 0..15
    const bf16* __restrict__ arow1 = sm.sbuf[16 + m16];  // M-tile 1: rows 16..31
    v8f acc[8] = {{}, {}, {}, {}, {}, {}, {}, {}};

    if (BF16W) {
      // fast path: W already bf16, K-padded -> uniform loop, B = two b128 loads
      for (int kb = 0; kb < KPAD; kb += 32) {
        V16 A0, A1;
        // A (contiguous k order): elems 0..15 = K kb+16*hi..kb+16*hi+15 (matches B)
        A0.u[0] = *(const uint4*)(arow0 + kb + 16 * hi);
        A0.u[1] = *(const uint4*)(arow0 + kb + 16 * hi + 8);
        A1.u[0] = *(const uint4*)(arow1 + kb + 16 * hi);
        A1.u[1] = *(const uint4*)(arow1 + kb + 16 * hi + 8);
        const int kcol = kb + 16 * hi;   // B: elems 0..15 = K kcol..kcol+15, column n0+m16
#pragma unroll
        for (int j = 0; j < 4; ++j) {
          const bf16* wr = Wbf + (size_t)(n0w + j * 16 + m16) * KPAD + kcol;
          V16 Bv;
          Bv.u[0] = *(const uint4*)(wr);
          Bv.u[1] = *(const uint4*)(wr + 8);
          acc[2 * j] = __builtin_amdgcn_wmma_f32_16x16x32_bf16(
              false, A0.v, false, Bv.v, (short)0, acc[2 * j], false, false);
          acc[2 * j + 1] = __builtin_amdgcn_wmma_f32_16x16x32_bf16(
              false, A1.v, false, Bv.v, (short)0, acc[2 * j + 1], false, false);
        }
      }
    } else {
      // fallback: f32 W converted on the fly
      for (int kb = 0; kb < 4672; kb += 32) {
        V16 A0, A1;
        A0.u[0] = *(const uint4*)(arow0 + kb + 16 * hi);
        A0.u[1] = *(const uint4*)(arow0 + kb + 16 * hi + 8);
        A1.u[0] = *(const uint4*)(arow1 + kb + 16 * hi);
        A1.u[1] = *(const uint4*)(arow1 + kb + 16 * hi + 8);
        const int kcol = kb + 16 * hi;
#pragma unroll
        for (int j = 0; j < 4; ++j) {
          const float* wr = W + (size_t)(n0w + j * 16 + m16) * SIGC + kcol;
          float4 b0 = *(const float4*)(wr);
          float4 b1 = *(const float4*)(wr + 4);
          float4 b2 = *(const float4*)(wr + 8);
          float4 b3 = *(const float4*)(wr + 12);
          v16bf Bv;
          Bv[0]  = (bf16)b0.x; Bv[1]  = (bf16)b0.y; Bv[2]  = (bf16)b0.z; Bv[3]  = (bf16)b0.w;
          Bv[4]  = (bf16)b1.x; Bv[5]  = (bf16)b1.y; Bv[6]  = (bf16)b1.z; Bv[7]  = (bf16)b1.w;
          Bv[8]  = (bf16)b2.x; Bv[9]  = (bf16)b2.y; Bv[10] = (bf16)b2.z; Bv[11] = (bf16)b2.w;
          Bv[12] = (bf16)b3.x; Bv[13] = (bf16)b3.y; Bv[14] = (bf16)b3.z; Bv[15] = (bf16)b3.w;
          acc[2 * j] = __builtin_amdgcn_wmma_f32_16x16x32_bf16(
              false, A0.v, false, Bv, (short)0, acc[2 * j], false, false);
          acc[2 * j + 1] = __builtin_amdgcn_wmma_f32_16x16x32_bf16(
              false, A1.v, false, Bv, (short)0, acc[2 * j + 1], false, false);
        }
      }
      { // tail K-step (kb = 4672): LDS pad is zero; guard W reads past 4680
        V16 A0, A1;
        A0.u[0] = *(const uint4*)(arow0 + 4672 + 16 * hi);
        A0.u[1] = *(const uint4*)(arow0 + 4680 + 16 * hi);
        A1.u[0] = *(const uint4*)(arow1 + 4672 + 16 * hi);
        A1.u[1] = *(const uint4*)(arow1 + 4680 + 16 * hi);
        const int kcol = 4672 + 16 * hi;
#pragma unroll
        for (int j = 0; j < 4; ++j) {
          const float* wr = W + (size_t)(n0w + j * 16 + m16) * SIGC;
          v16bf Bv;
#pragma unroll
          for (int t = 0; t < 16; ++t) {
            int kk = kcol + t;
            Bv[t] = (kk < SIGC) ? (bf16)wr[kk] : (bf16)0.0f;
          }
          acc[2 * j] = __builtin_amdgcn_wmma_f32_16x16x32_bf16(
              false, A0.v, false, Bv, (short)0, acc[2 * j], false, false);
          acc[2 * j + 1] = __builtin_amdgcn_wmma_f32_16x16x32_bf16(
              false, A1.v, false, Bv, (short)0, acc[2 * j + 1], false, false);
        }
      }
    }

    // ---------------- store C tiles (+bias), guard t == 255 ----------------
#pragma unroll
    for (int j = 0; j < 4; ++j) {
      const int n = n0w + j * 16 + m16;
      const float bv = bias[n];
#pragma unroll
      for (int mt = 0; mt < 2; ++mt) {
#pragma unroll
        for (int r = 0; r < 8; ++r) {
          const int t = chunk * MROWS + mt * 16 + r + 8 * hi;  // C layout: VGPR r, hi -> row r+8*hi
          if (t < NSTEP)
            out[((size_t)b * NSTEP + t) * ODIM + n] = acc[2 * j + mt][r] + bv;
        }
      }
    }
    __syncthreads();   // protect sbuf/dxc before next chunk overwrites them
  }
}

extern "C" void kernel_launch(void* const* d_in, const int* in_sizes, int n_in,
                              void* d_out, int out_size, void* d_ws, size_t ws_size,
                              hipStream_t stream) {
  const float* inp  = (const float*)d_in[0];   // [64,256,7]
  const float* W    = (const float*)d_in[1];   // [512,4680]
  const float* bias = (const float*)d_in[2];   // [512]
  float* out = (float*)d_out;                  // [64,255,512]
  (void)in_sizes; (void)n_in; (void)out_size;

  const size_t need = (size_t)ODIM * KPAD * sizeof(bf16);  // 4,816,896 B
  if (ws_size >= need) {
    bf16* Wbf = (bf16*)d_ws;
    convert_w_kernel<<<dim3(19, ODIM), 256, 0, stream>>>(W, Wbf);
    sig_stream_fused_kernel<true><<<64, 256, 0, stream>>>(inp, W, Wbf, bias, out);
  } else {
    sig_stream_fused_kernel<false><<<64, 256, 0, stream>>>(inp, W, nullptr, bias, out);
  }
}
